// LowPrecisionLeNet_47811575939758
// MI455X (gfx1250) — 8-bit matrix operand order repaired and checked by reading, compiled, not run on hardware
//
#include <hip/hip_runtime.h>
#include <hip/hip_bf16.h>

typedef __attribute__((ext_vector_type(8))) int v8i_t;

static constexpr size_t OFF_SCALES = 0;
static constexpr size_t OFF_BIAS   = 256;
static constexpr size_t OFF_WA_C1  = 1280;
static constexpr size_t OFF_WA_C2  = 3328;
static constexpr size_t OFF_WA_F1  = 6400;
static constexpr size_t OFF_WA_F2  = 63744;
static constexpr size_t OFF_WA_F3  = 76032;
static constexpr size_t OFF_A0     = 78336;
static constexpr size_t OFF_A1     = 50409984;
static constexpr size_t OFF_A2     = 69677568;
static constexpr size_t OFF_A3     = 76231168;
static constexpr size_t OFF_A4     = 78197248;

__device__ inline void write_A_blocks(int* dst, const float* w, float inv_s,
                                      int Mtot, int Ktot, int nMt, int nKc, int tid) {
  const int total = nMt * nKc * 256;
  for (int idx = tid; idx < total; idx += 256) {
    const int blk = idx >> 8, d = idx & 255;
    const int lane = d >> 3, v = d & 7;
    const int mt = blk / nKc, kc = blk % nKc;
    const int M = mt * 16 + (lane & 15);
    const int Kb = kc * 64 + ((v >> 2) << 5) + ((v & 3) << 2) + ((lane >> 4) << 4);
    unsigned int pack = 0;
#pragma unroll
    for (int b = 0; b < 4; ++b) {
      const int K = Kb + b;
      int q = 0;
      if (M < Mtot && K < Ktot) {
        q = (int)rintf(w[M * Ktot + K] * inv_s);
        q = q < -3 ? -3 : (q > 3 ? 3 : q);
      }
      pack |= (unsigned int)(q & 0xff) << (8 * b);
    }
    dst[idx] = pack;
  }
}

__global__ __launch_bounds__(256) void prep_kernel(
    const float* s_in, const float* w1, const float* b1, const float* s_r1,
    const float* w2, const float* b2, const float* s_r2,
    const float* wf1, const float* bf1, const float* s_r3,
    const float* wf2, const float* bf2, const float* s_r4,
    const float* wf3, unsigned char* ws) {
  __shared__ float red[256];
  __shared__ float sh_scale;
  const int tid = threadIdx.x;

  float* scales = (float*)(ws + OFF_SCALES);
  int* bias = (int*)(ws + OFF_BIAS);

  const float* wptr[5] = {w1, w2, wf1, wf2, wf3};
  const int wn[5] = {450, 2400, 48000, 10080, 840};
  float wsc[5];
  for (int t = 0; t < 5; ++t) {
    float m = 0.f;
    for (int i = tid; i < wn[t]; i += 256) m = fmaxf(m, fabsf(wptr[t][i]));
    red[tid] = m;
    __syncthreads();
    for (int s = 128; s > 0; s >>= 1) {
      if (tid < s) red[tid] = fmaxf(red[tid], red[tid + s]);
      __syncthreads();
    }
    if (tid == 0) sh_scale = fmaxf(red[0] * (1.0f / 3.0f), 1e-30f);
    __syncthreads();
    wsc[t] = sh_scale;
    __syncthreads();
  }

  write_A_blocks((int*)(ws + OFF_WA_C1), w1, 1.f / wsc[0], 6, 75, 1, 2, tid);
  write_A_blocks((int*)(ws + OFF_WA_C2), w2, 1.f / wsc[1], 16, 150, 1, 3, tid);
  write_A_blocks((int*)(ws + OFF_WA_F1), wf1, 1.f / wsc[2], 120, 400, 8, 7, tid);
  write_A_blocks((int*)(ws + OFF_WA_F2), wf2, 1.f / wsc[3], 84, 120, 6, 2, tid);
  write_A_blocks((int*)(ws + OFF_WA_F3), wf3, 1.f / wsc[4], 10, 84, 1, 2, tid);

  const float S1 = s_in[0] * wsc[0];
  const float S2 = s_r1[0] * wsc[1];
  const float S3 = s_r2[0] * wsc[2];
  const float S4 = s_r3[0] * wsc[3];
  if (tid < 6)   bias[tid]        = (int)rintf(b1[tid] / S1);
  if (tid < 16)  bias[16 + tid]   = (int)rintf(b2[tid] / S2);
  if (tid < 120) bias[32 + tid]   = (int)rintf(bf1[tid] / S3);
  if (tid < 84)  bias[160 + tid]  = (int)rintf(bf2[tid] / S4);
  if (tid == 0) {
    scales[0] = S1 / s_r1[0];
    scales[1] = S2 / s_r2[0];
    scales[2] = S3 / s_r3[0];
    scales[3] = S4 / s_r4[0];
    scales[4] = s_r4[0] * wsc[4];
  }
}

__global__ __launch_bounds__(256) void quant_input_kernel(
    const float* __restrict__ x, const float* __restrict__ s_in,
    unsigned int* __restrict__ a0) {
  const float inv = 1.0f / s_in[0];
  const size_t i = (size_t)blockIdx.x * 256 + threadIdx.x;
  const float4 v = ((const float4*)x)[i];
  int q0 = (int)rintf(v.x * inv); q0 = q0 < -8 ? -8 : (q0 > 7 ? 7 : q0);
  int q1 = (int)rintf(v.y * inv); q1 = q1 < -8 ? -8 : (q1 > 7 ? 7 : q1);
  int q2 = (int)rintf(v.z * inv); q2 = q2 < -8 ? -8 : (q2 > 7 ? 7 : q2);
  int q3 = (int)rintf(v.w * inv); q3 = q3 < -8 ? -8 : (q3 > 7 ? 7 : q3);
  a0[i] = (unsigned int)(q0 & 0xff) | ((unsigned int)(q1 & 0xff) << 8) |
          ((unsigned int)(q2 & 0xff) << 16) | ((unsigned int)(q3 & 0xff) << 24);
}

template <int KB>
__device__ inline void build_b_conv1(const unsigned char* __restrict__ q,
                                     v8i_t& bv0, v8i_t& bv1) {
#pragma unroll
  for (int v = 0; v < 8; ++v) {
    const int Ks = ((v >> 2) << 5) + KB + ((v & 3) << 2);
    unsigned int p0 = 0, p1 = 0;
#pragma unroll
    for (int bb = 0; bb < 4; ++bb) {
      const int K = Ks + bb;
      if (K < 75) {
        const int ch = K / 25, rr = K % 25, ky = rr / 5, kx = rr % 5;
        p0 |= (unsigned int)q[ch * 1024 + ky * 32 + kx] << (8 * bb);
      }
      const int K1 = K + 64;
      if (K1 < 75) {
        const int ch = K1 / 25, rr = K1 % 25, ky = rr / 5, kx = rr % 5;
        p1 |= (unsigned int)q[ch * 1024 + ky * 32 + kx] << (8 * bb);
      }
    }
    bv0[v] = (int)p0;
    bv1[v] = (int)p1;
  }
}

__global__ __launch_bounds__(256) void conv1_kernel(unsigned char* __restrict__ ws) {
  __shared__ unsigned char s_img[8 * 3072];
  __shared__ unsigned char s_pre[8 * 4704];
  const int tid = threadIdx.x, wave = tid >> 5, lane = tid & 31;
  const int img = blockIdx.x * 8 + wave;

  const unsigned int* a0 = (const unsigned int*)(ws + OFF_A0);
  unsigned int* img32 = (unsigned int*)s_img;
  for (int i = lane; i < 768; i += 32) img32[wave * 768 + i] = a0[(size_t)img * 768 + i];
  __syncthreads();

  const int* wA = (const int*)(ws + OFF_WA_C1);
  const v8i_t A0 = *(const v8i_t*)(wA + lane * 8);
  const v8i_t A1 = *(const v8i_t*)(wA + 256 + lane * 8);
  const float alpha = ((const float*)(ws + OFF_SCALES))[0];
  const int* bint = (const int*)(ws + OFF_BIAS);

  const unsigned char* im = s_img + wave * 3072;
  unsigned char* pre = s_pre + wave * 4704;
  const int n = lane & 15;

  int biasr[6];
#pragma unroll
  for (int r = 0; r < 6; ++r) biasr[r] = bint[r];

  for (int t = 0; t < 49; ++t) {
    const int p = t * 16 + n;
    const int oy = p / 28, ox = p % 28;
    const unsigned char* q = im + oy * 32 + ox;
    v8i_t bv0, bv1;
    if (lane < 16) build_b_conv1<0>(q, bv0, bv1);
    else           build_b_conv1<16>(q, bv0, bv1);
    v8i_t cacc = {};
    cacc = __builtin_amdgcn_wmma_i32_16x16x64_iu8(true, A0, true, bv0, cacc, false, false);
    cacc = __builtin_amdgcn_wmma_i32_16x16x64_iu8(true, A1, true, bv1, cacc, false, false);
    if (lane < 16) {
#pragma unroll
      for (int r = 0; r < 6; ++r) {
        const int acc = cacc[r] + biasr[r];
        int qv = (int)rintf((float)acc * alpha);
        qv = qv < 0 ? 0 : (qv > 15 ? 15 : qv);
        pre[r * 784 + p] = (unsigned char)qv;
      }
    }
  }
  __syncthreads();

  unsigned char* a1 = ws + OFF_A1;
  for (int i = lane; i < 1176; i += 32) {
    const int ch = i / 196, rr = i % 196, py = rr / 14, px = rr % 14;
    const unsigned char* q = pre + ch * 784 + (py * 2) * 28 + px * 2;
    unsigned char m = q[0];
    m = m > q[1] ? m : q[1];
    m = m > q[28] ? m : q[28];
    m = m > q[29] ? m : q[29];
    a1[(size_t)img * 1176 + i] = m;
  }
}

template <int KB>
__device__ inline void build_b_conv2(const unsigned char* __restrict__ q, v8i_t bv[3]) {
#pragma unroll
  for (int kc = 0; kc < 3; ++kc) {
#pragma unroll
    for (int v = 0; v < 8; ++v) {
      const int Ks = kc * 64 + ((v >> 2) << 5) + KB + ((v & 3) << 2);
      unsigned int pk = 0;
#pragma unroll
      for (int bb = 0; bb < 4; ++bb) {
        const int K = Ks + bb;
        if (K < 150) {
          const int ch = K / 25, rr = K % 25, ky = rr / 5, kx = rr % 5;
          pk |= (unsigned int)q[ch * 196 + ky * 14 + kx] << (8 * bb);
        }
      }
      bv[kc][v] = (int)pk;
    }
  }
}

__global__ __launch_bounds__(256) void conv2_kernel(unsigned char* __restrict__ ws) {
  __shared__ unsigned char s_in2[8 * 1184];
  __shared__ unsigned char s_pre[8 * 1600];
  const int tid = threadIdx.x, wave = tid >> 5, lane = tid & 31;
  const int img = blockIdx.x * 8 + wave;

  const unsigned char* a1 = ws + OFF_A1;
  {
    const unsigned int* src = (const unsigned int*)(a1 + (size_t)img * 1176);
    unsigned int* dst = (unsigned int*)(s_in2 + wave * 1184);
    for (int i = lane; i < 294; i += 32) dst[i] = src[i];
  }
  __syncthreads();

  const int* wA = (const int*)(ws + OFF_WA_C2);
  v8i_t Av[3];
#pragma unroll
  for (int k = 0; k < 3; ++k) Av[k] = *(const v8i_t*)(wA + k * 256 + lane * 8);
  const float alpha = ((const float*)(ws + OFF_SCALES))[1];
  const int* bint = (const int*)(ws + OFF_BIAS) + 16;

  const unsigned char* im = s_in2 + wave * 1184;
  unsigned char* pre = s_pre + wave * 1600;
  const int n = lane & 15;
  const int mhalf = (lane >> 4) << 3;

  int biasr[8];
#pragma unroll
  for (int r = 0; r < 8; ++r) biasr[r] = bint[r + mhalf];

  for (int t = 0; t < 7; ++t) {
    const int p = t * 16 + n;
    const int pv = p < 100 ? p : 99;
    const int oy = pv / 10, ox = pv % 10;
    const unsigned char* q = im + oy * 14 + ox;
    v8i_t bv[3];
    if (lane < 16) build_b_conv2<0>(q, bv);
    else           build_b_conv2<16>(q, bv);
    v8i_t cacc = {};
#pragma unroll
    for (int kc = 0; kc < 3; ++kc)
      cacc = __builtin_amdgcn_wmma_i32_16x16x64_iu8(true, Av[kc], true, bv[kc], cacc, false, false);
    if (p < 100) {
      unsigned char* prow = pre + mhalf * 100 + p;
#pragma unroll
      for (int r = 0; r < 8; ++r) {
        const int acc = cacc[r] + biasr[r];
        int qv = (int)rintf((float)acc * alpha);
        qv = qv < 0 ? 0 : (qv > 15 ? 15 : qv);
        prow[r * 100] = (unsigned char)qv;
      }
    }
  }
  __syncthreads();

  unsigned char* a2 = ws + OFF_A2;
  for (int i = lane; i < 400; i += 32) {
    const int ch = i / 25, rr = i % 25, py = rr / 5, px = rr % 5;
    const unsigned char* q = pre + ch * 100 + (py * 2) * 10 + px * 2;
    unsigned char m = q[0];
    m = m > q[1] ? m : q[1];
    m = m > q[10] ? m : q[10];
    m = m > q[11] ? m : q[11];
    a2[(size_t)img * 400 + i] = m;
  }
}

template <int KTOT, int KC, int MTOT>
__global__ __launch_bounds__(256) void fc_relu_kernel(
    unsigned char* __restrict__ ws, size_t in_off, size_t out_off, size_t wa_off,
    int scale_idx, int bias_off) {
  const int tid = threadIdx.x, wave = tid >> 5, lane = tid & 31;
  constexpr int nMt = (MTOT + 15) >> 4;
  if (wave >= nMt) return;
  const int mt = wave;
  const unsigned char* ain = ws + in_off;
  const int img = blockIdx.x * 16 + (lane & 15);
  const int kb = (lane >> 4) << 4;
  const int mhalf = (lane >> 4) << 3;
  const int* wA = (const int*)(ws + wa_off) + mt * KC * 256;

  const float alpha = ((const float*)(ws + OFF_SCALES))[scale_idx];
  const int* bint = (const int*)(ws + OFF_BIAS) + bias_off;
  const int mbase = mt * 16 + mhalf;
  int biasr[8];
#pragma unroll
  for (int r = 0; r < 8; ++r) biasr[r] = (mbase + r < MTOT) ? bint[mbase + r] : 0;

  const unsigned char* bbase = ain + (size_t)img * KTOT;
  v8i_t cacc = {};
#pragma unroll
  for (int kc = 0; kc < KC; ++kc) {
    const v8i_t A = *(const v8i_t*)(wA + kc * 256 + lane * 8);
    v8i_t bv;
#pragma unroll
    for (int v = 0; v < 8; ++v) {
      const int K = kc * 64 + ((v >> 2) << 5) + ((v & 3) << 2) + kb;
      int val = 0;
      if (K < KTOT) val = *(const int*)(bbase + K);
      bv[v] = val;
    }
    cacc = __builtin_amdgcn_wmma_i32_16x16x64_iu8(true, A, true, bv, cacc, false, false);
  }
  unsigned char* aout = ws + out_off + (size_t)img * MTOT;
  if (mbase + 7 < MTOT) {
#pragma unroll
    for (int r = 0; r < 8; ++r) {
      const int acc = cacc[r] + biasr[r];
      int qv = (int)rintf((float)acc * alpha);
      qv = qv < 0 ? 0 : (qv > 15 ? 15 : qv);
      aout[mbase + r] = (unsigned char)qv;
    }
  } else {
#pragma unroll
    for (int r = 0; r < 8; ++r) {
      if (mbase + r < MTOT) {
        const int acc = cacc[r] + biasr[r];
        int qv = (int)rintf((float)acc * alpha);
        qv = qv < 0 ? 0 : (qv > 15 ? 15 : qv);
        aout[mbase + r] = (unsigned char)qv;
      }
    }
  }
}

__global__ __launch_bounds__(256) void fc3_kernel(unsigned char* __restrict__ ws,
                                                  float* __restrict__ out) {
  const int tid = threadIdx.x, wave = tid >> 5, lane = tid & 31;
  const int ntile = blockIdx.x * 8 + wave;
  const int img = ntile * 16 + (lane & 15);
  const int kb = (lane >> 4) << 4;
  const unsigned char* ain = ws + OFF_A4;
  const int* wA = (const int*)(ws + OFF_WA_F3);
  const float sc = ((const float*)(ws + OFF_SCALES))[4];

  const unsigned char* bbase = ain + (size_t)img * 84;
  v8i_t cacc = {};
#pragma unroll
  for (int kc = 0; kc < 2; ++kc) {
    const v8i_t A = *(const v8i_t*)(wA + kc * 256 + lane * 8);
    v8i_t bv;
#pragma unroll
    for (int v = 0; v < 8; ++v) {
      const int K = kc * 64 + ((v >> 2) << 5) + ((v & 3) << 2) + kb;
      int val = 0;
      if (K < 84) val = *(const int*)(bbase + K);
      bv[v] = val;
    }
    cacc = __builtin_amdgcn_wmma_i32_16x16x64_iu8(true, A, true, bv, cacc, false, false);
  }
  float* orow = out + (size_t)img * 10;
  if (lane < 16) {
#pragma unroll
    for (int r = 0; r < 8; ++r) orow[r] = (float)cacc[r] * sc;
  } else {
#pragma unroll
    for (int r = 0; r < 2; ++r) orow[8 + r] = (float)cacc[r] * sc;
  }
}

extern "C" void kernel_launch(void* const* d_in, const int* in_sizes, int n_in,
                              void* d_out, int out_size, void* d_ws, size_t ws_size,
                              hipStream_t stream) {
  (void)in_sizes; (void)n_in; (void)out_size; (void)ws_size;
  const float* x    = (const float*)d_in[0];
  const float* s_in = (const float*)d_in[1];
  const float* w1   = (const float*)d_in[2];
  const float* b1   = (const float*)d_in[3];
  const float* s_r1 = (const float*)d_in[4];
  const float* w2   = (const float*)d_in[5];
  const float* b2   = (const float*)d_in[6];
  const float* s_r2 = (const float*)d_in[7];
  const float* wf1  = (const float*)d_in[8];
  const float* bf1  = (const float*)d_in[9];
  const float* s_r3 = (const float*)d_in[10];
  const float* wf2  = (const float*)d_in[11];
  const float* bf2  = (const float*)d_in[12];
  const float* s_r4 = (const float*)d_in[13];
  const float* wf3  = (const float*)d_in[14];
  unsigned char* ws = (unsigned char*)d_ws;
  float* out = (float*)d_out;

  prep_kernel<<<1, 256, 0, stream>>>(s_in, w1, b1, s_r1, w2, b2, s_r2,
                                     wf1, bf1, s_r3, wf2, bf2, s_r4, wf3, ws);
  quant_input_kernel<<<49152, 256, 0, stream>>>(x, s_in, (unsigned int*)(ws + OFF_A0));
  conv1_kernel<<<2048, 256, 0, stream>>>(ws);
  conv2_kernel<<<2048, 256, 0, stream>>>(ws);
  fc_relu_kernel<400, 7, 120><<<1024, 256, 0, stream>>>(ws, OFF_A2, OFF_A3, OFF_WA_F1, 2, 32);
  fc_relu_kernel<120, 2, 84><<<1024, 256, 0, stream>>>(ws, OFF_A3, OFF_A4, OFF_WA_F2, 3, 160);
  fc3_kernel<<<128, 256, 0, stream>>>(ws, out);
}
